// LSTM_LM_31920196944210
// MI455X (gfx1250) — matrix operand order repaired and checked by reading, not compiled, not run on hardware
//
#include <hip/hip_runtime.h>
#include <hip/hip_bf16.h>

typedef __attribute__((ext_vector_type(16))) _Float16 v16h;
typedef __attribute__((ext_vector_type(8)))  float    v8f;

#define B_    32
#define S_    40
#define L_    65
#define CE_   50
#define H_    50
#define DIN_  80
#define VW_   50000
#define VC_   256
#define CNN_  30
#define SM1_  39      // S_-1
#define ROWS_ 1248    // B_*(S_-1)

// ---------------- WMMA helpers (CDNA5 wave32 layouts) ----------------

__device__ __forceinline__ float sigf(float x) { return 1.f / (1.f + __expf(-x)); }

// A fragment (16x32 f16): lanes 0-15 hold row m=lane, K = {0..7, 16..23}+kBase,
// lanes 16-31 hold row m=lane-16, K = {8..15, 24..31}+kBase.
__device__ __forceinline__ v16h load_A_lds(const float* src, int ld, int kBase, int lane) {
  int m = lane & 15;
  int kOff = (lane & 16) ? 8 : 0;
  v16h a;
#pragma unroll
  for (int i = 0; i < 16; ++i) {
    int k = kBase + kOff + ((i < 8) ? i : (i + 8));
    a[i] = (_Float16)src[m * ld + k];
  }
  return a;
}

// D (16x16 f32): VGPR r holds row (r + 8*(lane>=16)), col = lane&15.
__device__ __forceinline__ void store_D_lds(float* dst, int ld, v8f d, int lane) {
  int n = lane & 15;
  int mBase = (lane & 16) ? 8 : 0;
#pragma unroll
  for (int r = 0; r < 8; ++r) dst[(mBase + r) * ld + n] = d[r];
}

__device__ __forceinline__ v8f wmma16(v16h a, v16h b, v8f c) {
  return __builtin_amdgcn_wmma_f32_16x16x32_f16(false, a, false, b, (short)0, c,
                                                false, false);
}

// ---------------- Pack B matrices into wave32 B-fragment layout ----------------
// Logical B[k][n] = Wcat[n][k], Wcat = [W1 | W2] with row-major (N x K1), (N x K2).
// Fragment layout: packed[((kt*NT + nt)*32 + lane)*16 + i] = B[kt*32 + 8*(lane/16) + (i<8 ? i : i+8)][nt*16 + lane%16].
__global__ void pack_B_kernel(const float* __restrict__ W1, int K1,
                              const float* __restrict__ W2, int K2,
                              int N, int KT, int NT,
                              _Float16* __restrict__ out) {
  int idx = blockIdx.x * blockDim.x + threadIdx.x;
  int total = KT * NT * 32;
  if (idx >= total) return;
  int lane = idx & 31;
  int tile = idx >> 5;
  int nt = tile % NT;
  int kt = tile / NT;
  int n = nt * 16 + (lane & 15);
  int K = K1 + K2;
  _Float16* dst = out + (size_t)idx * 16;
#pragma unroll 4
  for (int i = 0; i < 16; ++i) {
    int k = kt * 32 + ((lane & 16) ? 8 : 0) + ((i < 8) ? i : (i + 8));
    float v = 0.f;
    if (n < N && k < K)
      v = (k < K1) ? W1[(size_t)n * K1 + k] : W2[(size_t)n * K2 + (k - K1)];
    dst[i] = (_Float16)v;
  }
}

// ---------------- Embedding gather + char CNN (max-pooled) ----------------
// One block per (b,s). Output word_in[(b*S+s)*80]: [0..49]=word emb, [50..79]=cnn.
__global__ void __launch_bounds__(64)
embed_cnn_kernel(const int* __restrict__ wd, const int* __restrict__ cd,
                 const float* __restrict__ wembW, const float* __restrict__ cembW,
                 const float* __restrict__ convW, const float* __restrict__ convb,
                 float* __restrict__ word_in) {
  __shared__ int   ids[L_];
  __shared__ float emb[L_ * CE_];        // 3250
  __shared__ float wc[CNN_ * CE_ * 3];   // 4500
  __shared__ float y[63 * CNN_];         // 1890
  int bs = blockIdx.x;
  int tid = threadIdx.x;
  for (int i = tid; i < L_; i += 64) ids[i] = cd[bs * L_ + i];
  for (int i = tid; i < CNN_ * CE_ * 3; i += 64) wc[i] = convW[i];
  __syncthreads();
  for (int i = tid; i < L_ * CE_; i += 64) {
    int c = i / CE_, d = i % CE_;
    emb[i] = cembW[ids[c] * CE_ + d];
  }
  __syncthreads();
  if (tid < 63) {
    int p = tid;
    for (int ch = 0; ch < CNN_; ++ch) {
      float acc = 0.f;
#pragma unroll
      for (int w = 0; w < 3; ++w) {
        const float* e = &emb[(p + w) * CE_];
        const float* ww = &wc[ch * CE_ * 3 + w];
        for (int d = 0; d < CE_; ++d) acc += e[d] * ww[d * 3];
      }
      y[p * CNN_ + ch] = acc;
    }
  }
  __syncthreads();
  float* out = word_in + (size_t)bs * DIN_;
  if (tid < CNN_) {
    float m = -3.0e38f;
    for (int p = 0; p < 63; ++p) m = fmaxf(m, y[p * CNN_ + tid]);
    out[CE_ + tid] = m + convb[tid];
  }
  if (tid < CE_) out[tid] = wembW[(size_t)wd[bs] * CE_ + tid];
}

// ---------------- Word LSTM (single block, WMMA gates) ----------------
// gates(32x200) = [x(80) | h(50)](32x130) @ [Wih;Whh]^T ; K padded to 160 (KT=5), N to 208 (NT=13)
__global__ void __launch_bounds__(256)
word_lstm_kernel(const float* __restrict__ word_in, const int* __restrict__ mask,
                 const _Float16* __restrict__ packW,
                 const float* __restrict__ bih, const float* __restrict__ bhh,
                 float* __restrict__ word_rep) {
  __shared__ float xcat[32 * 160];
  __shared__ float gates[32 * 208];
  __shared__ float hs[32 * H_];
  __shared__ float cs[32 * H_];
  int tid = threadIdx.x, lane = tid & 31, wave = tid >> 5;
  for (int i = tid; i < 32 * H_; i += 256) { hs[i] = 0.f; cs[i] = 0.f; }
  for (int i = tid; i < 32 * 160; i += 256) xcat[i] = 0.f;
  __syncthreads();
  const v16h* bp = (const v16h*)packW;
  for (int s = 0; s < S_; ++s) {
    for (int i = tid; i < 32 * DIN_; i += 256) {
      int r = i / DIN_, j = i % DIN_;
      xcat[r * 160 + j] = word_in[((size_t)r * S_ + s) * DIN_ + j];
    }
    for (int i = tid; i < 32 * H_; i += 256) {
      int r = i / H_, j = i % H_;
      xcat[r * 160 + DIN_ + j] = hs[i];
    }
    __syncthreads();
    for (int tile = wave; tile < 26; tile += 8) {
      int mt = tile & 1, nt = tile >> 1;
      v8f acc = {0.f, 0.f, 0.f, 0.f, 0.f, 0.f, 0.f, 0.f};
#pragma unroll
      for (int kt = 0; kt < 5; ++kt) {
        v16h a = load_A_lds(&xcat[mt * 16 * 160], 160, kt * 32, lane);
        v16h b = bp[(kt * 13 + nt) * 32 + lane];
        acc = wmma16(a, b, acc);
      }
      store_D_lds(&gates[mt * 16 * 208 + nt * 16], 208, acc, lane);
    }
    __syncthreads();
    for (int i = tid; i < 32 * H_; i += 256) {
      int r = i / H_, j = i % H_;
      const float* g = &gates[r * 208];
      float gi = g[j]           + bih[j]           + bhh[j];
      float gf = g[H_ + j]      + bih[H_ + j]      + bhh[H_ + j];
      float gg = g[2 * H_ + j]  + bih[2 * H_ + j]  + bhh[2 * H_ + j];
      float go = g[3 * H_ + j]  + bih[3 * H_ + j]  + bhh[3 * H_ + j];
      float c = sigf(gf) * cs[i] + sigf(gi) * tanhf(gg);
      float h = sigf(go) * tanhf(c);
      cs[i] = c; hs[i] = h;
      float mk = (float)mask[r * S_ + s];
      word_rep[((size_t)r * S_ + s) * H_ + j] = h * mk;
    }
    __syncthreads();
  }
}

// ---------------- Word classifier + online log-softmax NLL ----------------
// 78 blocks x 16 rows; stream 3125 N-tiles of wcls with online softmax.
__global__ void __launch_bounds__(256)
word_cls_kernel(const float* __restrict__ word_rep, const int* __restrict__ wd,
                const _Float16* __restrict__ packW, const float* __restrict__ wb,
                float* __restrict__ word_nll) {
  __shared__ float arow[16 * 64];
  __shared__ int   targ[16];
  __shared__ float tlog[16];
  __shared__ float pm[16 * 128];
  __shared__ float ps[16 * 128];
  int tid = threadIdx.x, lane = tid & 31, wave = tid >> 5;
  int rowBase = blockIdx.x * 16;
  for (int i = tid; i < 16 * 64; i += 256) arow[i] = 0.f;
  __syncthreads();
  for (int i = tid; i < 16 * H_; i += 256) {
    int r = i / H_, j = i % H_;
    int rg = rowBase + r;
    int b = rg / SM1_, s = rg % SM1_;
    arow[r * 64 + j] = word_rep[((size_t)b * S_ + s) * H_ + j];
  }
  if (tid < 16) {
    int rg = rowBase + tid;
    int b = rg / SM1_, s = rg % SM1_;
    targ[tid] = wd[b * S_ + s + 1];
    tlog[tid] = 0.f;
  }
  __syncthreads();
  v16h a0 = load_A_lds(arow, 64, 0, lane);
  v16h a1 = load_A_lds(arow, 64, 32, lane);
  const v16h* bp = (const v16h*)packW;
  float mx[8], se[8];
#pragma unroll
  for (int r = 0; r < 8; ++r) { mx[r] = -3.0e38f; se[r] = 0.f; }
  int nCol = lane & 15;
  int mBase = (lane & 16) ? 8 : 0;
  for (int nt = wave; nt < 3125; nt += 8) {
    v8f acc = {0.f, 0.f, 0.f, 0.f, 0.f, 0.f, 0.f, 0.f};
    acc = wmma16(a0, bp[(0 * 3125 + nt) * 32 + lane], acc);
    acc = wmma16(a1, bp[(1 * 3125 + nt) * 32 + lane], acc);
    int n = nt * 16 + nCol;
    float bias = wb[n];
#pragma unroll
    for (int r = 0; r < 8; ++r) {
      float x = acc[r] + bias;
      int row = mBase + r;
      if (n == targ[row]) tlog[row] = x;   // unique (lane,wave) per (row,n)
      float nm = fmaxf(mx[r], x);
      se[r] = se[r] * __expf(mx[r] - nm) + __expf(x - nm);
      mx[r] = nm;
    }
  }
  int slot = wave * 16 + nCol;
#pragma unroll
  for (int r = 0; r < 8; ++r) {
    pm[(mBase + r) * 128 + slot] = mx[r];
    ps[(mBase + r) * 128 + slot] = se[r];
  }
  __syncthreads();
  if (tid < 16) {
    float m = -3.0e38f;
    for (int i = 0; i < 128; ++i) m = fmaxf(m, pm[tid * 128 + i]);
    float s = 0.f;
    for (int i = 0; i < 128; ++i) s += ps[tid * 128 + i] * __expf(pm[tid * 128 + i] - m);
    float lse = m + __logf(s);
    word_nll[rowBase + tid] = (targ[tid] == 0) ? 0.f : (lse - tlog[tid]);
  }
}

// ---------------- Char LSTM fused with char classifier NLL ----------------
// 78 blocks x 16 sequences; 65 steps; gates GEMM (K=100->KT=4, NT=13) and
// 256-way classifier GEMM (K=50->KT=2, NT=16) per step, log-softmax in LDS.
__global__ void __launch_bounds__(256)
char_lstm_kernel(const float* __restrict__ word_rep, const int* __restrict__ cd,
                 const float* __restrict__ cembW,
                 const _Float16* __restrict__ packG,
                 const float* __restrict__ cbih, const float* __restrict__ cbhh,
                 const _Float16* __restrict__ packC, const float* __restrict__ ccb,
                 float* __restrict__ char_nll) {
  __shared__ float xcat[16 * 128];
  __shared__ float gates[16 * 208];
  __shared__ float hbuf[16 * 64];
  __shared__ float cbuf[16 * H_];
  __shared__ float logits[16 * 256];
  __shared__ float pmx[16 * 16];
  __shared__ float pse[16 * 16];
  __shared__ float nllsum[16];
  int tid = threadIdx.x, lane = tid & 31, wave = tid >> 5;
  int rowBase = blockIdx.x * 16;
  for (int i = tid; i < 16 * 128; i += 256) xcat[i] = 0.f;
  for (int i = tid; i < 16 * 64; i += 256) hbuf[i] = 0.f;
  for (int i = tid; i < 16 * H_; i += 256) cbuf[i] = 0.f;
  if (tid < 16) nllsum[tid] = 0.f;
  __syncthreads();
  const v16h* gp = (const v16h*)packG;
  const v16h* cp = (const v16h*)packC;
  for (int t = 0; t < L_; ++t) {
    // x: word_rep at t==0, else char embedding of previous char
    for (int i = tid; i < 16 * H_; i += 256) {
      int r = i / H_, j = i % H_;
      int rg = rowBase + r;
      int b = rg / SM1_, s = rg % SM1_;
      float x;
      if (t == 0) x = word_rep[((size_t)b * S_ + s) * H_ + j];
      else {
        int cid = cd[((size_t)b * S_ + s + 1) * L_ + (t - 1)];
        x = cembW[(size_t)cid * CE_ + j];
      }
      xcat[r * 128 + j] = x;
      xcat[r * 128 + H_ + j] = hbuf[r * 64 + j];
    }
    __syncthreads();
    for (int nt = wave; nt < 13; nt += 8) {
      v8f acc = {0.f, 0.f, 0.f, 0.f, 0.f, 0.f, 0.f, 0.f};
#pragma unroll
      for (int kt = 0; kt < 4; ++kt) {
        v16h a = load_A_lds(xcat, 128, kt * 32, lane);
        acc = wmma16(a, gp[(kt * 13 + nt) * 32 + lane], acc);
      }
      store_D_lds(&gates[nt * 16], 208, acc, lane);
    }
    __syncthreads();
    for (int i = tid; i < 16 * H_; i += 256) {
      int r = i / H_, j = i % H_;
      const float* g = &gates[r * 208];
      float gi = g[j]          + cbih[j]          + cbhh[j];
      float gf = g[H_ + j]     + cbih[H_ + j]     + cbhh[H_ + j];
      float gg = g[2 * H_ + j] + cbih[2 * H_ + j] + cbhh[2 * H_ + j];
      float go = g[3 * H_ + j] + cbih[3 * H_ + j] + cbhh[3 * H_ + j];
      float c = sigf(gf) * cbuf[i] + sigf(gi) * tanhf(gg);
      float h = sigf(go) * tanhf(c);
      cbuf[i] = c;
      hbuf[r * 64 + j] = h;
    }
    __syncthreads();
    for (int nt = wave; nt < 16; nt += 8) {
      v8f acc = {0.f, 0.f, 0.f, 0.f, 0.f, 0.f, 0.f, 0.f};
#pragma unroll
      for (int kt = 0; kt < 2; ++kt) {
        v16h a = load_A_lds(hbuf, 64, kt * 32, lane);
        acc = wmma16(a, cp[(kt * 16 + nt) * 32 + lane], acc);
      }
      int n = nt * 16 + (lane & 15);
      int mB = (lane & 16) ? 8 : 0;
      float bias = ccb[n];
#pragma unroll
      for (int r = 0; r < 8; ++r) logits[(mB + r) * 256 + n] = acc[r] + bias;
    }
    __syncthreads();
    { // partial max / sum-exp over 16-col chunks
      int r = tid >> 4;
      int c0 = (tid & 15) * 16;
      float m = -3.0e38f;
      for (int i = 0; i < 16; ++i) m = fmaxf(m, logits[r * 256 + c0 + i]);
      float s = 0.f;
      for (int i = 0; i < 16; ++i) s += __expf(logits[r * 256 + c0 + i] - m);
      pmx[r * 16 + (tid & 15)] = m;
      pse[r * 16 + (tid & 15)] = s;
    }
    __syncthreads();
    if (tid < 16) {
      int rg = rowBase + tid;
      int b = rg / SM1_, s = rg % SM1_;
      int tg = cd[((size_t)b * S_ + s + 1) * L_ + t];
      float m = -3.0e38f;
      for (int i = 0; i < 16; ++i) m = fmaxf(m, pmx[tid * 16 + i]);
      float se = 0.f;
      for (int i = 0; i < 16; ++i) se += pse[tid * 16 + i] * __expf(pmx[tid * 16 + i] - m);
      float lse = m + __logf(se);
      if (tg != 0) nllsum[tid] += lse - logits[tid * 256 + tg];
    }
    __syncthreads();
  }
  if (tid < 16) char_nll[rowBase + tid] = nllsum[tid];
}

// ---------------- Final mixture / losses / reductions ----------------
__global__ void __launch_bounds__(256)
final_kernel(const float* __restrict__ word_rep, const int* __restrict__ mask,
             const float* __restrict__ smW, const float* __restrict__ smb,
             const float* __restrict__ word_nll, const float* __restrict__ char_nll,
             float* __restrict__ out) {
  __shared__ float sent[B_];
  __shared__ int   len[B_];
  __shared__ float r1[256], r2[256], r3[256];
  int tid = threadIdx.x;
  if (tid < B_) {
    sent[tid] = 0.f;
    int c = 0;
    for (int s = 0; s < S_; ++s) c += mask[tid * S_ + s];
    len[tid] = c - 1;
  }
  __syncthreads();
  float lc = 0.f, l2 = 0.f, lp2 = 0.f;
  for (int rg = tid; rg < ROWS_; rg += 256) {
    int b = rg / SM1_, s = rg % SM1_;
    const float* wr = &word_rep[((size_t)b * S_ + s) * H_];
    float acc = smb[0];
    for (int j = 0; j < H_; ++j) acc += wr[j] * smW[j];
    float g = sigf(acc);
    float wp = __expf(-word_nll[rg]);
    float cpb = __expf(-char_nll[rg]);
    float twp = (1.f - g) * wp + g * cpb;
    atomicAdd(&sent[b], __logf(twp));
    lc += cpb;
    if (s < len[b]) {
      float lg2 = log2f(twp);
      l2 += lg2;
      lp2 += twp * lg2;
    }
  }
  r1[tid] = lc; r2[tid] = l2; r3[tid] = lp2;
  __syncthreads();
  for (int off = 128; off > 0; off >>= 1) {
    if (tid < off) {
      r1[tid] += r1[tid + off];
      r2[tid] += r2[tid + off];
      r3[tid] += r3[tid + off];
    }
    __syncthreads();
  }
  if (tid < B_) out[2 + tid] = sent[tid];
  if (tid == 0) {
    float tot = 0.f;
    for (int b = 0; b < B_; ++b) tot += sent[b];
    out[0] = -tot / (float)B_;          // loss
    out[1] = r1[0] / (float)ROWS_;      // mean char_prob
    out[34] = r2[0];                    // sum_log2
    out[35] = r3[0];                    // sum_p_log2
  }
}

// ---------------- Host orchestration ----------------
extern "C" void kernel_launch(void* const* d_in, const int* in_sizes, int n_in,
                              void* d_out, int out_size, void* d_ws, size_t ws_size,
                              hipStream_t stream) {
  const int*   wd    = (const int*)d_in[0];
  const int*   cd    = (const int*)d_in[1];
  const int*   mask  = (const int*)d_in[2];
  const float* wembW = (const float*)d_in[3];
  const float* cembW = (const float*)d_in[4];
  const float* convW = (const float*)d_in[5];
  const float* convb = (const float*)d_in[6];
  const float* lWih  = (const float*)d_in[7];
  const float* lWhh  = (const float*)d_in[8];
  const float* lbih  = (const float*)d_in[9];
  const float* lbhh  = (const float*)d_in[10];
  const float* cWih  = (const float*)d_in[11];
  const float* cWhh  = (const float*)d_in[12];
  const float* cbih  = (const float*)d_in[13];
  const float* cbhh  = (const float*)d_in[14];
  const float* wclsW = (const float*)d_in[15];
  const float* wclsb = (const float*)d_in[16];
  const float* cclsW = (const float*)d_in[17];
  const float* cclsb = (const float*)d_in[18];
  const float* smW   = (const float*)d_in[19];
  const float* smb   = (const float*)d_in[20];

  char* ws = (char*)d_ws;
  size_t o = 0;
  auto alloc = [&](size_t bytes) { size_t cur = o; o += (bytes + 255) & ~(size_t)255; return cur; };

  const size_t nWcls  = (size_t)2 * 3125 * 32 * 16; // halves
  const size_t nCcls  = (size_t)2 * 16   * 32 * 16;
  const size_t nWlstm = (size_t)5 * 13   * 32 * 16;
  const size_t nClstm = (size_t)4 * 13   * 32 * 16;

  _Float16* p_wcls  = (_Float16*)(ws + alloc(nWcls  * sizeof(_Float16)));
  _Float16* p_ccls  = (_Float16*)(ws + alloc(nCcls  * sizeof(_Float16)));
  _Float16* p_wlstm = (_Float16*)(ws + alloc(nWlstm * sizeof(_Float16)));
  _Float16* p_clstm = (_Float16*)(ws + alloc(nClstm * sizeof(_Float16)));
  float* word_in  = (float*)(ws + alloc((size_t)B_ * S_ * DIN_ * sizeof(float)));
  float* word_rep = (float*)(ws + alloc((size_t)B_ * S_ * H_ * sizeof(float)));
  float* word_nll = (float*)(ws + alloc((size_t)ROWS_ * sizeof(float)));
  float* char_nll = (float*)(ws + alloc((size_t)ROWS_ * sizeof(float)));

  // 1) pack B-fragment layouts
  {
    int tot = 2 * 3125 * 32;
    pack_B_kernel<<<(tot + 255) / 256, 256, 0, stream>>>(wclsW, 50, nullptr, 0, VW_, 2, 3125, p_wcls);
  }
  {
    int tot = 2 * 16 * 32;
    pack_B_kernel<<<(tot + 255) / 256, 256, 0, stream>>>(cclsW, 50, nullptr, 0, VC_, 2, 16, p_ccls);
  }
  {
    int tot = 5 * 13 * 32;
    pack_B_kernel<<<(tot + 255) / 256, 256, 0, stream>>>(lWih, 80, lWhh, 50, 200, 5, 13, p_wlstm);
  }
  {
    int tot = 4 * 13 * 32;
    pack_B_kernel<<<(tot + 255) / 256, 256, 0, stream>>>(cWih, 50, cWhh, 50, 200, 4, 13, p_clstm);
  }

  // 2) embeddings + char CNN
  embed_cnn_kernel<<<B_ * S_, 64, 0, stream>>>(wd, cd, wembW, cembW, convW, convb, word_in);

  // 3) word LSTM
  word_lstm_kernel<<<1, 256, 0, stream>>>(word_in, mask, p_wlstm, lbih, lbhh, word_rep);

  // 4) word classifier + NLL
  word_cls_kernel<<<ROWS_ / 16, 256, 0, stream>>>(word_rep, wd, p_wcls, wclsb, word_nll);

  // 5) char LSTM fused with char classifier NLL
  char_lstm_kernel<<<ROWS_ / 16, 256, 0, stream>>>(word_rep, cd, cembW, p_clstm, cbih, cbhh,
                                                   p_ccls, cclsb, char_nll);

  // 6) final combine
  final_kernel<<<1, 256, 0, stream>>>(word_rep, mask, smW, smb, word_nll, char_nll, (float*)d_out);
}
